// EEGEncoder_rwkv_v1_60129542390
// MI455X (gfx1250) — compile-verified, matrix operand order checked by reading, not run on hardware
//
#include <hip/hip_runtime.h>
#include <hip/hip_bf16.h>

typedef __attribute__((ext_vector_type(16))) _Float16 v16h;
typedef __attribute__((ext_vector_type(8)))  float    v8f;
typedef unsigned int u32x4 __attribute__((ext_vector_type(4)));
typedef int          i32x4 __attribute__((ext_vector_type(4)));
typedef int          i32x8 __attribute__((ext_vector_type(8)));

#if defined(__has_builtin)
#  if __has_builtin(__builtin_amdgcn_tensor_load_to_lds) && __has_builtin(__builtin_amdgcn_s_wait_tensorcnt)
#    define HAVE_TDM 1
#  endif
#endif
#ifndef HAVE_TDM
#  define HAVE_TDM 0
#endif

#define EPSN 1e-5f

__device__ __forceinline__ float gelu_exact(float x) {
    return 0.5f * x * (1.0f + erff(x * 0.70710678118654752f));
}
__device__ __forceinline__ float sigm(float x) {
    return 1.0f / (1.0f + expf(-x));
}

// ---------------------------------------------------------------------------
// K0: build fused graph matrix M[32x32] = W0^T + W1^T L + W2^T L^2, rowsum(M)
// ---------------------------------------------------------------------------
__global__ void k_graph_setup(const float* __restrict__ A, const float* __restrict__ gcw,
                              float* __restrict__ Mout, float* __restrict__ RSout) {
    __shared__ float Ar[32][32];
    __shared__ float Lm[32][33];
    __shared__ float L2[32][33];
    __shared__ float Ms[32][32];
    __shared__ float dinv[32];
    int tid = threadIdx.x;          // blockDim.x == 1024
    int i = tid >> 5, j = tid & 31;
    float a = A[i * 32 + j];
    Ar[i][j] = a > 0.0f ? a : 0.0f;
    __syncthreads();
    if (j == 0) {
        float s = 0.0f;
        for (int c = 0; c < 32; ++c) s += Ar[i][c];
        dinv[i] = rsqrtf(s + 1e-10f);
    }
    __syncthreads();
    Lm[i][j] = dinv[i] * Ar[i][j] * dinv[j];
    __syncthreads();
    float acc = 0.0f;
    for (int c = 0; c < 32; ++c) acc += Lm[i][c] * Lm[c][j];
    L2[i][j] = acc;
    __syncthreads();
    // M[o][j], o==i
    float m = gcw[j * 32 + i];  // W0[j,o]
    for (int c = 0; c < 32; ++c) {
        m += gcw[1024 + c * 32 + i] * Lm[c][j];
        m += gcw[2048 + c * 32 + i] * L2[c][j];
    }
    Ms[i][j] = m;
    Mout[i * 32 + j] = m;
    __syncthreads();
    if (j == 0) {
        float s = 0.0f;
        for (int c = 0; c < 32; ++c) s += Ms[i][c];
        RSout[i] = s;
    }
}

// ---------------------------------------------------------------------------
// K1: fused per-t BN + graph conv + ReLU; write f16 activations (b,32,T0)
// ---------------------------------------------------------------------------
__global__ void k_graph_apply(const float* __restrict__ X,
                              const float* __restrict__ g, const float* __restrict__ bb,
                              const float* __restrict__ mm, const float* __restrict__ vv,
                              const float* __restrict__ Mbuf, const float* __restrict__ RS,
                              _Float16* __restrict__ Y, int T0) {
    __shared__ float Msh[1024];
    __shared__ float RSsh[32];
    int tid = threadIdx.x;
    for (int t2 = tid; t2 < 1024; t2 += 256) Msh[t2] = Mbuf[t2];
    if (tid < 32) RSsh[tid] = RS[tid];
    __syncthreads();
    int t = blockIdx.x * 256 + tid;
    int b = blockIdx.y;
    if (t >= T0) return;
    float scale = g[t] * rsqrtf(vv[t] + EPSN);
    float shift = bb[t] - mm[t] * scale;
    float acc[32];
#pragma unroll
    for (int o = 0; o < 32; ++o) acc[o] = 0.0f;
    const float* xb = X + (size_t)b * 32 * T0 + t;
    for (int j = 0; j < 32; ++j) {
        float xv = xb[(size_t)j * T0];
        __builtin_prefetch(xb + (size_t)(j + 1) * T0, 0, 1);
#pragma unroll
        for (int o = 0; o < 32; ++o) acc[o] += Msh[o * 32 + j] * xv;
    }
    _Float16* yb = Y + (size_t)b * 32 * T0 + t;
#pragma unroll
    for (int o = 0; o < 32; ++o) {
        float r = scale * acc[o] + shift * RSsh[o];
        r = r > 0.0f ? r : 0.0f;
        yb[(size_t)o * T0] = (_Float16)r;
    }
}

// ---------------------------------------------------------------------------
// K2: proj conv (k=8, stride=4, 32->64 ch) as WMMA GEMM: W(64x256) x patches.
// B tile (32 ci-rows x 72 halves, pitch T0) staged global->LDS via the Tensor
// Data Mover (D# per CDNA5 ISA), then fed from ds loads.
// grid = (456, 4, 24), block = 32 (one wave per 16x16 tile)
// ---------------------------------------------------------------------------
__global__ void k_projconv_wmma(const _Float16* __restrict__ G,
                                const _Float16* __restrict__ Wh,
                                float* __restrict__ Y) {
    const int T0 = 29184, L1 = 7295;
    __shared__ __align__(16) _Float16 shB[32 * 72];
    int lt = blockIdx.x, ot = blockIdx.y, b = blockIdx.z;
    int lane = threadIdx.x;
    int half = lane >> 4, lm = lane & 15;
    int l0 = lt * 16;
    const _Float16* gtile = G + (size_t)b * 32 * T0 + 4 * l0;

#if HAVE_TDM
    {
        unsigned long long gaddr = (unsigned long long)(const char*)gtile;
        unsigned int ldsoff = (unsigned int)(unsigned long long)(size_t)(void*)&shB[0];
        unsigned int dim0 = (unsigned int)(T0 - 4 * l0);  // remaining row -> OOB reads 0
        // D# group 0: count=1 | lds_addr | global_addr[56:0] | type=2
        u32x4 g0 = { 1u, ldsoff,
                     (unsigned int)(gaddr & 0xFFFFFFFFu),
                     (unsigned int)(((gaddr >> 32) & 0x01FFFFFFu) | (2u << 30)) };
        // D# group 1: data_size=2B; tensor_dim0=dim0; tensor_dim1=32;
        //             tile_dim0=72, tile_dim1=32; tensor_dim0_stride=T0
        i32x8 g1 = { (int)0x00010000,
                     (int)((dim0 & 0xFFFFu) << 16),
                     (int)(((dim0 >> 16) & 0xFFFFu) | (32u << 16)),
                     (int)(72u << 16),
                     (int)32,
                     (int)T0,
                     0, 0 };
        i32x4 z4 = { 0, 0, 0, 0 };
#if defined(__clang_major__) && __clang_major__ >= 23
        i32x8 z8 = { 0, 0, 0, 0, 0, 0, 0, 0 };
        __builtin_amdgcn_tensor_load_to_lds(g0, g1, z4, z4, z8, 0);
#else
        __builtin_amdgcn_tensor_load_to_lds(g0, g1, z4, z4, 0);
#endif
        __builtin_amdgcn_s_wait_tensorcnt(0);
    }
#else
    {   // fallback: cooperative global->LDS stage (zero-fill OOB tail)
        int base_t = 4 * l0;
        for (int idx = lane; idx < 32 * 36; idx += 32) {
            int row = idx / 36, c4 = idx % 36;
            int tcol = c4 * 2;
            unsigned int val = 0;
            if (base_t + tcol + 1 < T0)
                val = *(const unsigned int*)(gtile + (size_t)row * T0 + tcol);
            *(unsigned int*)(&shB[row * 72 + tcol]) = val;
        }
        __syncthreads();
    }
#endif

    v8f acc = {};
    union UV { unsigned int u[8]; v16h h; };
#pragma unroll
    for (int kc = 0; kc < 8; ++kc) {
        int kbase = kc * 32;
        UV Au, Bu;
#pragma unroll
        for (int vv2 = 0; vv2 < 8; ++vv2) {
            int k0 = (vv2 < 4) ? (half * 8 + 2 * vv2) : (16 + half * 8 + 2 * (vv2 - 4));
            int k = kbase + k0;
            Au.u[vv2] = *(const unsigned int*)(Wh + (size_t)(ot * 16 + lm) * 256 + k);
            int ci = k >> 3, kk = k & 7;
            Bu.u[vv2] = *(const unsigned int*)(&shB[ci * 72 + 4 * lm + kk]);
        }
        acc = __builtin_amdgcn_wmma_f32_16x16x32_f16(false, Au.h, false, Bu.h,
                                                     (short)0, acc, false, false);
    }
    int lo = l0 + lm;
#pragma unroll
    for (int r2 = 0; r2 < 8; ++r2) {
        int co = ot * 16 + r2 + 8 * half;
        if (lo < L1) Y[((size_t)b * 64 + co) * L1 + lo] = acc[r2];
    }
}

// ---------------------------------------------------------------------------
// channel LayerNorm over strided channel dim, in place: X (B, C, L)
// ---------------------------------------------------------------------------
__global__ void k_chan_ln(float* __restrict__ X, const float* __restrict__ g,
                          const float* __restrict__ bt, int B, int C, int L) {
    int i = blockIdx.x * blockDim.x + threadIdx.x;
    if (i >= B * L) return;
    int l = i % L, b = i / L;
    float* p = X + (size_t)b * C * L + l;
    float mu = 0.0f;
    for (int c = 0; c < C; ++c) mu += p[(size_t)c * L];
    mu /= (float)C;
    float var = 0.0f;
    for (int c = 0; c < C; ++c) { float d = p[(size_t)c * L] - mu; var += d * d; }
    var /= (float)C;
    float inv = rsqrtf(var + EPSN);
    for (int c = 0; c < C; ++c)
        p[(size_t)c * L] = (p[(size_t)c * L] - mu) * inv * g[c] + bt[c];
}

// ---------------------------------------------------------------------------
// 1x1 conv (+ optional bias): Y[b,o,l] = sum_c W[o,c] X[b,c,l] + bias[o]
// ---------------------------------------------------------------------------
__global__ void k_conv1x1(const float* __restrict__ X, const float* __restrict__ W,
                          const float* __restrict__ bias, float* __restrict__ Y,
                          int B, int Cin, int Cout, int L) {
    int i = blockIdx.x * blockDim.x + threadIdx.x;
    int total = B * Cout * L;
    if (i >= total) return;
    int l = i % L; int rest = i / L; int o = rest % Cout; int b = rest / Cout;
    const float* xp = X + (size_t)b * Cin * L + l;
    const float* wp = W + (size_t)o * Cin;
    float acc = bias ? bias[o] : 0.0f;
    for (int c = 0; c < Cin; ++c) acc += wp[c] * xp[(size_t)c * L];
    Y[i] = acc;
}

// res-block stage 1: y1 = prelu(bn(W1 @ x), a)
__global__ void k_rb_conv_bn_prelu(const float* __restrict__ X, const float* __restrict__ W,
                                   const float* __restrict__ g, const float* __restrict__ bt,
                                   const float* __restrict__ mn, const float* __restrict__ vr,
                                   const float* __restrict__ aP, float* __restrict__ Y,
                                   int B, int Cin, int Cout, int L) {
    int i = blockIdx.x * blockDim.x + threadIdx.x;
    int total = B * Cout * L;
    if (i >= total) return;
    int l = i % L; int rest = i / L; int o = rest % Cout; int b = rest / Cout;
    const float* xp = X + (size_t)b * Cin * L + l;
    const float* wp = W + (size_t)o * Cin;
    float acc = 0.0f;
    for (int c = 0; c < Cin; ++c) acc += wp[c] * xp[(size_t)c * L];
    float sc = g[o] * rsqrtf(vr[o] + EPSN);
    float val = (acc - mn[o]) * sc + bt[o];
    float a = aP[0];
    Y[i] = val >= 0.0f ? val : a * val;
}

// res-block tail: z = prelu(bn2(W2@y1) + shortcut, a2); out = maxpool3(z)
__global__ void k_rb_tail(const float* __restrict__ Y1, const float* __restrict__ X,
                          const float* __restrict__ W2,
                          const float* __restrict__ g, const float* __restrict__ bt,
                          const float* __restrict__ mn, const float* __restrict__ vr,
                          const float* __restrict__ a2P, const float* __restrict__ Wd,
                          float* __restrict__ Y,
                          int B, int Cin, int Cout, int Lin, int Lout) {
    int i = blockIdx.x * blockDim.x + threadIdx.x;
    int total = B * Cout * Lout;
    if (i >= total) return;
    int lo = i % Lout; int rest = i / Lout; int o = rest % Cout; int b = rest / Cout;
    float sc = g[o] * rsqrtf(vr[o] + EPSN);
    float a2 = a2P[0];
    float best = -3.4e38f;
    for (int ll = 0; ll < 3; ++ll) {
        int l = lo * 3 + ll;
        const float* yp = Y1 + (size_t)b * Cout * Lin + l;
        float acc = 0.0f;
        for (int c = 0; c < Cout; ++c) acc += W2[(size_t)o * Cout + c] * yp[(size_t)c * Lin];
        float val = (acc - mn[o]) * sc + bt[o];
        float s;
        if (Wd) {
            const float* xp = X + (size_t)b * Cin * Lin + l;
            s = 0.0f;
            for (int c = 0; c < Cin; ++c) s += Wd[(size_t)o * Cin + c] * xp[(size_t)c * Lin];
        } else {
            s = X[((size_t)b * Cin + o) * Lin + l];
        }
        val += s;
        val = val >= 0.0f ? val : a2 * val;
        best = fmaxf(best, val);
    }
    Y[i] = best;
}

// ---------------------------------------------------------------------------
// f32 -> f16 weight conversion
// ---------------------------------------------------------------------------
__global__ void k_f32tof16(const float* __restrict__ s, _Float16* __restrict__ d, int n) {
    int i = blockIdx.x * blockDim.x + threadIdx.x;
    if (i < n) d[i] = (_Float16)s[i];
}

// gather (b,c,t) -> rows (b*t, c)
__global__ void k_bt_gather(const float* __restrict__ I, float* __restrict__ X0) {
    int i = blockIdx.x * blockDim.x + threadIdx.x;
    if (i >= 24 * 270 * 64) return;
    int c = i % 64; int t = (i / 64) % 270; int b = i / (64 * 270);
    X0[i] = I[((size_t)b * 64 + c) * 270 + t];
}

// ---------------------------------------------------------------------------
// WMMA f16 GEMM, K compile-time (fully unrolled), 16x32 tile per wave:
// Y[n,o] = act( sum_k X[n,k]*W[o,k] + bias[o] )
// act: 0 none, 1 gelu, 2 sigmoid, 3 relu^2. N%16==0, O%32==0.
// grid = (N/16, O/32), block = 32
// ---------------------------------------------------------------------------
template <int K>
__global__ void k_gemm_wmma_t(const float* __restrict__ X, const _Float16* __restrict__ Wh,
                              const float* __restrict__ bias, float* __restrict__ Y,
                              int N, int O, int act) {
    int nt = blockIdx.x, og = blockIdx.y;
    int lane = threadIdx.x;
    int half = lane >> 4, lm = lane & 15;
    v8f acc0 = {}, acc1 = {};
    union HV { _Float16 e[16]; v16h h; };
    const float*    xrow = X  + (size_t)(nt * 16 + lm) * K;
    const _Float16* w0   = Wh + (size_t)(og * 32 + lm) * K;
    const _Float16* w1   = Wh + (size_t)(og * 32 + 16 + lm) * K;
#pragma unroll
    for (int kb = 0; kb < K; kb += 32) {
        HV Au, B0, B1;
#pragma unroll
        for (int vv2 = 0; vv2 < 8; ++vv2) {
            int k0 = (vv2 < 4) ? (half * 8 + 2 * vv2) : (16 + half * 8 + 2 * (vv2 - 4));
            int k = kb + k0;
            Au.e[2 * vv2]     = (_Float16)xrow[k];
            Au.e[2 * vv2 + 1] = (_Float16)xrow[k + 1];
            B0.e[2 * vv2]     = w0[k];
            B0.e[2 * vv2 + 1] = w0[k + 1];
            B1.e[2 * vv2]     = w1[k];
            B1.e[2 * vv2 + 1] = w1[k + 1];
        }
        acc0 = __builtin_amdgcn_wmma_f32_16x16x32_f16(false, Au.h, false, B0.h,
                                                      (short)0, acc0, false, false);
        acc1 = __builtin_amdgcn_wmma_f32_16x16x32_f16(false, Au.h, false, B1.h,
                                                      (short)0, acc1, false, false);
    }
    int o0 = og * 32 + lm;
    int o1 = o0 + 16;
    float bs0 = bias ? bias[o0] : 0.0f;
    float bs1 = bias ? bias[o1] : 0.0f;
#pragma unroll
    for (int r2 = 0; r2 < 8; ++r2) {
        int n = nt * 16 + r2 + 8 * half;
        float v0 = acc0[r2] + bs0;
        float v1 = acc1[r2] + bs1;
        if (act == 1)      { v0 = gelu_exact(v0); v1 = gelu_exact(v1); }
        else if (act == 2) { v0 = sigm(v0);       v1 = sigm(v1); }
        else if (act == 3) { v0 = v0 > 0.0f ? v0 * v0 : 0.0f;
                             v1 = v1 > 0.0f ? v1 * v1 : 0.0f; }
        Y[(size_t)n * O + o0] = v0;
        Y[(size_t)n * O + o1] = v1;
    }
}

// row LayerNorm over contiguous C (in-place safe)
__global__ void k_ln_rows(const float* __restrict__ X, const float* __restrict__ g,
                          const float* __restrict__ bt, float* __restrict__ Y, int N, int C) {
    int n = blockIdx.x * blockDim.x + threadIdx.x;
    if (n >= N) return;
    const float* xp = X + (size_t)n * C;
    float mu = 0.0f;
    for (int c = 0; c < C; ++c) mu += xp[c];
    mu /= (float)C;
    float var = 0.0f;
    for (int c = 0; c < C; ++c) { float d = xp[c] - mu; var += d * d; }
    var /= (float)C;
    float inv = rsqrtf(var + EPSN);
    float* yp = Y + (size_t)n * C;
    for (int c = 0; c < C; ++c) yp[c] = (xp[c] - mu) * inv * g[c] + bt[c];
}

// ---------------------------------------------------------------------------
// bidirectional WKV: one thread per (b, channel); fwd scan stores carries,
// bwd scan combines on the fly. k,v,y layout: (b, t, C), C fastest.
// ---------------------------------------------------------------------------
__global__ void k_biwkv(const float* __restrict__ K_, const float* __restrict__ V_,
                        const float* __restrict__ decay, const float* __restrict__ first,
                        float* __restrict__ pf, float* __restrict__ af, float* __restrict__ bf,
                        float* __restrict__ Yb, int B, int T, int C) {
    int idx = blockIdx.x * blockDim.x + threadIdx.x;
    if (idx >= B * C) return;
    int b = idx / C, d = idx % C;
    float w = -expf(decay[d] / (float)T);
    float u = first[d] / (float)T;
    const float* kp = K_ + (size_t)b * T * C + d;
    const float* vp = V_ + (size_t)b * T * C + d;
    float* pfp = pf + (size_t)b * T * C + d;
    float* afp = af + (size_t)b * T * C + d;
    float* bfp = bf + (size_t)b * T * C + d;
    float* yp  = Yb + (size_t)b * T * C + d;
    float p = -1e30f, a = 0.0f, bb2 = 0.0f;
    for (int t = 0; t < T; ++t) {
        size_t off = (size_t)t * C;
        pfp[off] = p; afp[off] = a; bfp[off] = bb2;
        float kt = kp[off], vt = vp[off];
        float q  = fmaxf(p + w, kt);
        float e1 = expf(p + w - q), e2 = expf(kt - q);
        a = e1 * a + e2 * vt; bb2 = e1 * bb2 + e2; p = q;
    }
    p = -1e30f; a = 0.0f; bb2 = 0.0f;
    for (int t = T - 1; t >= 0; --t) {
        size_t off = (size_t)t * C;
        float kt = kp[off], vt = vp[off];
        float s = u + kt;
        float pff = pfp[off], aff = afp[off], bff = bfp[off];
        float m  = fmaxf(fmaxf(pff, p), s);
        float ef = expf(pff - m), eb = expf(p - m), es = expf(s - m);
        yp[off] = (ef * aff + eb * a + es * vt) / (ef * bff + eb * bb2 + es);
        float q  = fmaxf(p + w, kt);
        float e1 = expf(p + w - q), e2 = expf(kt - q);
        a = e1 * a + e2 * vt; bb2 = e1 * bb2 + e2; p = q;
    }
}

__global__ void k_ew_mul(float* __restrict__ D, const float* __restrict__ S, int n) {
    int i = blockIdx.x * blockDim.x + threadIdx.x;
    if (i < n) D[i] *= S[i];
}
__global__ void k_ew_addto(float* __restrict__ D, const float* __restrict__ S, int n) {
    int i = blockIdx.x * blockDim.x + threadIdx.x;
    if (i < n) D[i] += S[i];
}
__global__ void k_ew_add2(const float* __restrict__ A, const float* __restrict__ Bs,
                          float* __restrict__ D, int n) {
    int i = blockIdx.x * blockDim.x + threadIdx.x;
    if (i < n) D[i] = A[i] + Bs[i];
}

// final: out[b,c,t] = identity[b,c,t] + h2[(b*T+t)*64 + c]
__global__ void k_final(const float* __restrict__ I, const float* __restrict__ H2,
                        float* __restrict__ out) {
    int i = blockIdx.x * blockDim.x + threadIdx.x;
    if (i >= 24 * 64 * 270) return;
    int t = i % 270; int c = (i / 270) % 64; int b = i / (270 * 64);
    out[i] = I[i] + H2[((size_t)b * 270 + t) * 64 + c];
}

// ===========================================================================
extern "C" void kernel_launch(void* const* d_in, const int* in_sizes, int n_in,
                              void* d_out, int out_size, void* d_ws, size_t ws_size,
                              hipStream_t stream) {
    (void)in_sizes; (void)n_in; (void)out_size; (void)ws_size;
    const int B = 24, T0 = 29184, L1 = 7295, LP1 = 2431, LP2 = 810, LP3 = 270;
    const int Dm = 128, Hm = 256, T = 270, N = B * T;  // 6480
    (void)T;

    // ---- inputs (setup_inputs dict flattened in insertion order) ----
    const float* spike = (const float*)d_in[0];
    const float* bn1g = (const float*)d_in[1];
    const float* bn1b = (const float*)d_in[2];
    const float* bn1m = (const float*)d_in[3];
    const float* bn1v = (const float*)d_in[4];
    const float* Amat = (const float*)d_in[5];
    const float* gcw  = (const float*)d_in[6];
    const float* projw= (const float*)d_in[7];
    const float* clng = (const float*)d_in[8];
    const float* clnb = (const float*)d_in[9];
    const float* inw  = (const float*)d_in[10];
    const float* inb  = (const float*)d_in[11];
    const float* r1w1 = (const float*)d_in[12];
    const float* r1g1 = (const float*)d_in[13];
    const float* r1b1 = (const float*)d_in[14];
    const float* r1m1 = (const float*)d_in[15];
    const float* r1v1 = (const float*)d_in[16];
    const float* r1a1 = (const float*)d_in[17];
    const float* r1w2 = (const float*)d_in[18];
    const float* r1g2 = (const float*)d_in[19];
    const float* r1b2 = (const float*)d_in[20];
    const float* r1m2 = (const float*)d_in[21];
    const float* r1v2 = (const float*)d_in[22];
    const float* r1a2 = (const float*)d_in[23];
    const float* r2w1 = (const float*)d_in[24];
    const float* r2g1 = (const float*)d_in[25];
    const float* r2b1 = (const float*)d_in[26];
    const float* r2m1 = (const float*)d_in[27];
    const float* r2v1 = (const float*)d_in[28];
    const float* r2a1 = (const float*)d_in[29];
    const float* r2w2 = (const float*)d_in[30];
    const float* r2g2 = (const float*)d_in[31];
    const float* r2b2 = (const float*)d_in[32];
    const float* r2m2 = (const float*)d_in[33];
    const float* r2v2 = (const float*)d_in[34];
    const float* r2a2 = (const float*)d_in[35];
    const float* r2wd = (const float*)d_in[36];
    const float* r3w1 = (const float*)d_in[37];
    const float* r3g1 = (const float*)d_in[38];
    const float* r3b1 = (const float*)d_in[39];
    const float* r3m1 = (const float*)d_in[40];
    const float* r3v1 = (const float*)d_in[41];
    const float* r3a1 = (const float*)d_in[42];
    const float* r3w2 = (const float*)d_in[43];
    const float* r3g2 = (const float*)d_in[44];
    const float* r3b2 = (const float*)d_in[45];
    const float* r3m2 = (const float*)d_in[46];
    const float* r3v2 = (const float*)d_in[47];
    const float* r3a2 = (const float*)d_in[48];
    const float* outw = (const float*)d_in[49];
    const float* outb = (const float*)d_in[50];
    const float* l1w  = (const float*)d_in[51];
    const float* l1b  = (const float*)d_in[52];
    const float* ln0g = (const float*)d_in[53];
    const float* ln0b = (const float*)d_in[54];
    const float* ln1g = (const float*)d_in[55];
    const float* ln1b = (const float*)d_in[56];
    const float* ln2g = (const float*)d_in[57];
    const float* ln2b = (const float*)d_in[58];
    const float* awk  = (const float*)d_in[59];
    const float* awv  = (const float*)d_in[60];
    const float* awr  = (const float*)d_in[61];
    const float* awo  = (const float*)d_in[62];
    const float* adec = (const float*)d_in[63];
    const float* afir = (const float*)d_in[64];
    const float* akng = (const float*)d_in[65];
    const float* aknb = (const float*)d_in[66];
    const float* fwk  = (const float*)d_in[67];
    const float* fwr  = (const float*)d_in[68];
    const float* fwv  = (const float*)d_in[69];
    const float* fkng = (const float*)d_in[70];
    const float* fknb = (const float*)d_in[71];
    const float* l2w  = (const float*)d_in[72];
    const float* l2b  = (const float*)d_in[73];
    float* out = (float*)d_out;

    // ---- workspace layout (lifetime-based slab reuse) ----
    char* wsb = (char*)d_ws;
    float* SLAB0 = (float*)(wsb + 0);           // 44,820,480 B
    float* SLAB1 = (float*)(wsb + 44820480);    // 44,820,480 B
    float* SLAB2 = (float*)(wsb + 89640960);    // 14,936,064 B
    _Float16* G16 = (_Float16*)(wsb + 104577024);  // 44,826,624 B
    float* IDB   = (float*)(wsb + 149403648);   // 1,658,880 B
    float* Mbuf  = (float*)(wsb + 151062528);   // 4 KB
    float* RSbuf = (float*)(wsb + 151066624);
    _Float16* W16 = (_Float16*)(wsb + 151066880);
    _Float16* Wproj16 = W16;
    _Float16* Wl1_16  = Wproj16 + 16384;
    _Float16* Wk16    = Wl1_16 + 8192;
    _Float16* Wv16    = Wk16 + 16384;
    _Float16* Wr16    = Wv16 + 16384;
    _Float16* Wo16    = Wr16 + 16384;
    _Float16* Wfk16   = Wo16 + 16384;
    _Float16* Wfr16   = Wfk16 + 32768;
    _Float16* Wfv16   = Wfr16 + 16384;
    _Float16* Wl2_16  = Wfv16 + 32768;

    // stage-F buffers (overlap SLAB0/SLAB1 region — dead by then)
    float* fX0  = (float*)(wsb + 0);
    float* fh0  = (float*)(wsb + 1658880);
    float* fxa  = (float*)(wsb + 4976640);
    float* fk   = (float*)(wsb + 8294400);
    float* fv   = (float*)(wsb + 11612160);
    float* fr   = (float*)(wsb + 14929920);
    float* fpf  = (float*)(wsb + 18247680);
    float* faf  = (float*)(wsb + 21565440);
    float* fbf  = (float*)(wsb + 24883200);
    float* fy   = (float*)(wsb + 28200960);
    float* fsm  = (float*)(wsb + 31518720);
    float* fxatt= (float*)(wsb + 34836480);
    float* fkffn= (float*)(wsb + 38154240);
    float* fcm  = (float*)(wsb + 44789760);
    float* frf  = (float*)(wsb + 48107520);
    float* fh2  = (float*)(wsb + 51425280);

    auto blocks = [](int n) { return (n + 255) / 256; };
    auto cvt = [&](const float* s, _Float16* d, int n) {
        k_f32tof16<<<blocks(n), 256, 0, stream>>>(s, d, n);
    };

    // ---- weight conversions to f16 ----
    cvt(projw, Wproj16, 16384);
    cvt(l1w, Wl1_16, 8192);
    cvt(awk, Wk16, 16384);  cvt(awv, Wv16, 16384);
    cvt(awr, Wr16, 16384);  cvt(awo, Wo16, 16384);
    cvt(fwk, Wfk16, 32768); cvt(fwr, Wfr16, 16384); cvt(fwv, Wfv16, 32768);
    cvt(l2w, Wl2_16, 8192);

    // ---- front pipeline ----
    k_graph_setup<<<1, 1024, 0, stream>>>(Amat, gcw, Mbuf, RSbuf);
    k_graph_apply<<<dim3(T0 / 256, B), 256, 0, stream>>>(spike, bn1g, bn1b, bn1m, bn1v,
                                                         Mbuf, RSbuf, G16, T0);
    k_projconv_wmma<<<dim3(456, 4, B), 32, 0, stream>>>(G16, Wproj16, SLAB0);
    k_chan_ln<<<blocks(B * L1), 256, 0, stream>>>(SLAB0, clng, clnb, B, 64, L1);
    k_conv1x1<<<blocks(B * 64 * L1), 256, 0, stream>>>(SLAB0, inw, inb, SLAB1, B, 64, 64, L1);

    // rb1 (64->64, L 7295 -> 2431)
    k_rb_conv_bn_prelu<<<blocks(B * 64 * L1), 256, 0, stream>>>(
        SLAB1, r1w1, r1g1, r1b1, r1m1, r1v1, r1a1, SLAB0, B, 64, 64, L1);
    k_rb_tail<<<blocks(B * 64 * LP1), 256, 0, stream>>>(
        SLAB0, SLAB1, r1w2, r1g2, r1b2, r1m2, r1v2, r1a2, nullptr, SLAB2,
        B, 64, 64, L1, LP1);
    // rb2 (64->32, L 2431 -> 810, with downsample)
    k_rb_conv_bn_prelu<<<blocks(B * 32 * LP1), 256, 0, stream>>>(
        SLAB2, r2w1, r2g1, r2b1, r2m1, r2v1, r2a1, SLAB0, B, 64, 32, LP1);
    k_rb_tail<<<blocks(B * 32 * LP2), 256, 0, stream>>>(
        SLAB0, SLAB2, r2w2, r2g2, r2b2, r2m2, r2v2, r2a2, r2wd, SLAB1,
        B, 64, 32, LP1, LP2);
    // rb3 (32->32, L 810 -> 270)
    k_rb_conv_bn_prelu<<<blocks(B * 32 * LP2), 256, 0, stream>>>(
        SLAB1, r3w1, r3g1, r3b1, r3m1, r3v1, r3a1, SLAB0, B, 32, 32, LP2);
    k_rb_tail<<<blocks(B * 32 * LP3), 256, 0, stream>>>(
        SLAB0, SLAB1, r3w2, r3g2, r3b2, r3m2, r3v2, r3a2, nullptr, SLAB2,
        B, 32, 32, LP2, LP3);
    // out conv -> identity buffer (b,64,270)
    k_conv1x1<<<blocks(B * 64 * LP3), 256, 0, stream>>>(SLAB2, outw, outb, IDB, B, 32, 64, LP3);

    // ---- stage F: token MLP + RWKV (N = 6480 rows) ----
    const int NE = N * Dm;  // 829,440
    k_bt_gather<<<blocks(N * 64), 256, 0, stream>>>(IDB, fX0);
    k_gemm_wmma_t<64><<<dim3(N / 16, 128 / 32), 32, 0, stream>>>(fX0, Wl1_16, l1b, fh0, N, 128, 1);
    k_ln_rows<<<blocks(N), 256, 0, stream>>>(fh0, ln0g, ln0b, fxa, N, Dm);  // ln0
    k_gemm_wmma_t<128><<<dim3(N / 16, 128 / 32), 32, 0, stream>>>(fxa, Wk16, nullptr, fk, N, 128, 0);
    k_gemm_wmma_t<128><<<dim3(N / 16, 128 / 32), 32, 0, stream>>>(fxa, Wv16, nullptr, fv, N, 128, 0);
    k_gemm_wmma_t<128><<<dim3(N / 16, 128 / 32), 32, 0, stream>>>(fxa, Wr16, nullptr, fr, N, 128, 2);
    k_biwkv<<<blocks(B * Dm), 256, 0, stream>>>(fk, fv, adec, afir,
                                                fpf, faf, fbf, fy, B, 270, Dm);
    k_ln_rows<<<blocks(N), 256, 0, stream>>>(fy, akng, aknb, fy, N, Dm);    // kn LN
    k_ew_mul<<<blocks(NE), 256, 0, stream>>>(fy, fr, NE);    // sr * y
    k_gemm_wmma_t<128><<<dim3(N / 16, 128 / 32), 32, 0, stream>>>(fy, Wo16, nullptr, fsm, N, 128, 0);
    k_ln_rows<<<blocks(N), 256, 0, stream>>>(fsm, ln1g, ln1b, fsm, N, Dm);  // ln1
    k_ew_add2<<<blocks(NE), 256, 0, stream>>>(fxa, fsm, fxatt, NE);         // x += ln1(att)
    // channel mix
    k_gemm_wmma_t<128><<<dim3(N / 16, 256 / 32), 32, 0, stream>>>(fxatt, Wfk16, nullptr, fkffn, N, 256, 3);
    k_ln_rows<<<blocks(N), 256, 0, stream>>>(fkffn, fkng, fknb, fkffn, N, Hm);
    k_gemm_wmma_t<128><<<dim3(N / 16, 128 / 32), 32, 0, stream>>>(fxatt, Wfr16, nullptr, frf, N, 128, 2);
    k_gemm_wmma_t<256><<<dim3(N / 16, 128 / 32), 32, 0, stream>>>(fkffn, Wfv16, nullptr, fcm, N, 128, 0);
    k_ew_mul<<<blocks(NE), 256, 0, stream>>>(fcm, frf, NE);
    k_ln_rows<<<blocks(N), 256, 0, stream>>>(fcm, ln2g, ln2b, fcm, N, Dm);  // ln2
    k_ew_addto<<<blocks(NE), 256, 0, stream>>>(fxatt, fcm, NE);             // x += ln2(cm)
    k_gemm_wmma_t<128><<<dim3(N / 16, 64 / 32), 32, 0, stream>>>(fxatt, Wl2_16, l2b, fh2, N, 64, 1);
    k_final<<<blocks(B * 64 * LP3), 256, 0, stream>>>(IDB, fh2, out);
}
